// ProteinMPNN_33440615367146
// MI455X (gfx1250) — matrix operand order repaired and checked by reading, not compiled, not run on hardware
//
#include <hip/hip_runtime.h>
#include <hip/hip_bf16.h>

// ---------------- problem constants ----------------
#define Bdim   4
#define Ldim   1024
#define Kdim   48
#define Hdim   128
#define FFdim  512
#define KIN    384          // 3*H
#define ROWS   128          // rows per workgroup in edge kernels
#define LN_EPS 1e-5f
#define SCALE_INV (1.0f/30.0f)

typedef __attribute__((ext_vector_type(16))) __bf16 v16bf;
typedef __attribute__((ext_vector_type(8)))  float  v8f;
typedef unsigned int u32x4 __attribute__((ext_vector_type(4)));
typedef unsigned int u32x8 __attribute__((ext_vector_type(8)));
typedef int          i32x4 __attribute__((ext_vector_type(4)));
typedef int          i32x8 __attribute__((ext_vector_type(8)));

// global-address-space qualified types (force global_load_* instead of flat_load_*)
typedef const __attribute__((address_space(1))) u32x8    g_u32x8;
typedef const __attribute__((address_space(1))) float    g_f32;
typedef const __attribute__((address_space(1))) int      g_i32;

static __device__ __forceinline__ float ldgf(const float* p) {
  return *(g_f32*)(unsigned long long)p;
}
static __device__ __forceinline__ int ldgi(const int* p) {
  return *(g_i32*)(unsigned long long)p;
}

// ---------------- small helpers ----------------
static __device__ __forceinline__ float gelu_f(float x) {
  return 0.5f * x * (1.0f + erff(x * 0.70710678118654752f));
}

// A fragment: 16x32 bf16, row-major LDS source with stride lda.
// Per lane this is two contiguous 16B chunks -> 2x ds_load_b128, free unpack.
static __device__ __forceinline__ v16bf load_a_frag(const __bf16* A, int m0, int k0,
                                                    int lda, int lane) {
  int m = lane & 15, g = lane >> 4;
  const __bf16* base = A + (m0 + m) * lda + k0;
  u32x4 u0 = *(const u32x4*)(base + g * 16);       // K = g*16 .. g*16+7
  u32x4 u1 = *(const u32x4*)(base + g * 16 + 8);   // K = g*16+8 .. g*16+15
  u32x8 u = __builtin_shufflevector(u0, u1, 0, 1, 2, 3, 4, 5, 6, 7);
  return __builtin_bit_cast(v16bf, u);
}

// B fragment: 32x16 bf16 from weights pre-transposed to [N][K] (global bf16).
// Per lane: 32 contiguous bytes -> global_load_b128 x2, free unpack.
static __device__ __forceinline__ v16bf load_b_frag(const __bf16* Wt, int n0, int k0,
                                                    int ldk, int lane) {
  int n = lane & 15, g = lane >> 4;
  const __bf16* base = Wt + (size_t)(n0 + n) * ldk + k0 + g * 16;
  u32x8 u = *(g_u32x8*)(unsigned long long)base;
  return __builtin_bit_cast(v16bf, u);
}

static __device__ __forceinline__ v8f wmma_bf16(v16bf a, v16bf b, v8f c) {
  return __builtin_amdgcn_wmma_f32_16x16x32_bf16(false, a, false, b, (short)0, c,
                                                 false, false);
}

// ---------------- Tensor Data Mover: 1-D f32 tile global -> LDS ----------------
// lds_off: byte offset inside the workgroup's dynamic-LDS allocation (base 0).
// nelem:   number of f32 elements (must fit in 16-bit tile_dim0).
static __device__ __forceinline__ void tdm_load_f32(unsigned lds_off, const void* gptr,
                                                    unsigned nelem) {
  unsigned long long ga = (unsigned long long)gptr;
  u32x4 g0 = {};
  g0[0] = 1u;                                            // count=1, user descriptor
  g0[1] = lds_off;                                       // lds_addr
  g0[2] = (unsigned)ga;                                  // global_addr[31:0]
  g0[3] = (unsigned)((ga >> 32) & 0x01ffffffu) | (2u << 30);  // [56:32] | type=2
  i32x8 g1 = {};
  g1[0] = (2 << 16);                                     // data_size = 4 bytes
  g1[1] = (int)((nelem & 0xffffu) << 16);                // tensor_dim0[15:0]
  g1[2] = (int)((nelem >> 16) & 0xffffu) | (1 << 16);    // tensor_dim0[31:16], tensor_dim1=1
  g1[3] = (int)((nelem & 0xffffu) << 16);                // tile_dim0
  g1[4] = 1;                                             // tile_dim1 = 1
  g1[5] = (int)nelem;                                    // tensor_dim0_stride[31:0]
  i32x4 gz = {};
#if defined(__clang_major__) && (__clang_major__ >= 23)
  i32x8 gz8 = {};
  __builtin_amdgcn_tensor_load_to_lds(g0, g1, gz, gz, gz8, 0);
#else
  __builtin_amdgcn_tensor_load_to_lds(g0, g1, gz, gz, 0);
#endif
}

// ---------------- fused 3-layer MLP core (per 128-row tile) ----------------
// Each wave owns 16 rows (m0 = wave*16): gelu(gelu(A@W1+b1)@W2+b2)@W3 -> out[8].
// B fragments are double-buffered so a global load is in flight across each WMMA.
template <int KC>
static __device__ __forceinline__ void mlp3_core(const __bf16* Abuf,
                                                 const __bf16* w1t, const float* b1,
                                                 const __bf16* w2t, const float* b2,
                                                 const __bf16* w3t,
                                                 __bf16* act1, __bf16* act2,
                                                 v8f out[8], int lane, int wave,
                                                 bool syncA) {
  const int m0 = wave * 16;
  const int nidx = lane & 15, g = lane >> 4;
  constexpr int NKT1 = KC / 32;

  // ---- layer 1 ----
  v16bf af[NKT1];
#pragma unroll
  for (int kt = 0; kt < NKT1; ++kt) af[kt] = load_a_frag(Abuf, m0, kt * 32, KC, lane);
  if (syncA) __syncthreads();

#pragma unroll
  for (int nt = 0; nt < 8; ++nt) {
    v8f acc = {};
    v16bf bc_ = load_b_frag(w1t, nt * 16, 0, KC, lane);
#pragma unroll
    for (int kt = 0; kt < NKT1; ++kt) {
      v16bf bn = bc_;
      if (kt + 1 < NKT1) bn = load_b_frag(w1t, nt * 16, (kt + 1) * 32, KC, lane);
      acc = wmma_bf16(af[kt], bc_, acc);
      bc_ = bn;
    }
    int col = nt * 16 + nidx;
    float bcv = ldgf(b1 + col);
#pragma unroll
    for (int j = 0; j < 8; ++j) {
      int row = m0 + j + g * 8;
      act1[row * Hdim + col] = (__bf16)gelu_f(acc[j] + bcv);
    }
  }

  // ---- layer 2 (wave reads only its own rows -> no barrier needed) ----
  v16bf af2[4];
#pragma unroll
  for (int kt = 0; kt < 4; ++kt) af2[kt] = load_a_frag(act1, m0, kt * 32, Hdim, lane);
#pragma unroll
  for (int nt = 0; nt < 8; ++nt) {
    v8f acc = {};
    v16bf bc_ = load_b_frag(w2t, nt * 16, 0, Hdim, lane);
#pragma unroll
    for (int kt = 0; kt < 4; ++kt) {
      v16bf bn = bc_;
      if (kt + 1 < 4) bn = load_b_frag(w2t, nt * 16, (kt + 1) * 32, Hdim, lane);
      acc = wmma_bf16(af2[kt], bc_, acc);
      bc_ = bn;
    }
    int col = nt * 16 + nidx;
    float bcv = ldgf(b2 + col);
#pragma unroll
    for (int j = 0; j < 8; ++j) {
      int row = m0 + j + g * 8;
      act2[row * Hdim + col] = (__bf16)gelu_f(acc[j] + bcv);
    }
  }

  // ---- layer 3 ----
  v16bf af3[4];
#pragma unroll
  for (int kt = 0; kt < 4; ++kt) af3[kt] = load_a_frag(act2, m0, kt * 32, Hdim, lane);
#pragma unroll
  for (int nt = 0; nt < 8; ++nt) {
    v8f acc = {};
    v16bf bc_ = load_b_frag(w3t, nt * 16, 0, Hdim, lane);
#pragma unroll
    for (int kt = 0; kt < 4; ++kt) {
      v16bf bn = bc_;
      if (kt + 1 < 4) bn = load_b_frag(w3t, nt * 16, (kt + 1) * 32, Hdim, lane);
      acc = wmma_bf16(af3[kt], bc_, acc);
      bc_ = bn;
    }
    out[nt] = acc;
  }
}

// ---------------- prep kernels ----------------
__global__ void k_zero(float* p, int n) {
  int i = blockIdx.x * 256 + threadIdx.x;
  if (i < n) p[i] = 0.0f;
}
// W: [K][N] f32 row-major  ->  Wt: [N][K] bf16
__global__ void k_wt(const float* __restrict__ W, __bf16* __restrict__ Wt, int K, int N) {
  int i = blockIdx.x * 256 + threadIdx.x;
  if (i < K * N) {
    int k = i / N, n = i - k * N;
    Wt[(size_t)n * K + k] = (__bf16)ldgf(W + i);
  }
}

// ---------------- kernel 1: node-message MLP + masked scatter-sum ----------------
__global__ __launch_bounds__(256) void k_edge_node(
    const float* __restrict__ hV, const float* __restrict__ hE,
    const float* __restrict__ mask_attend, const int* __restrict__ Eidx,
    const __bf16* __restrict__ w1t, const float* __restrict__ b1,
    const __bf16* __restrict__ w2t, const float* __restrict__ b2,
    const __bf16* __restrict__ w3t, const float* __restrict__ b3,
    float* __restrict__ dh) {
  extern __shared__ char smem[];
  __bf16* Abuf = (__bf16*)smem;                         // 128*384*2 = 96KB
  __bf16* act1 = (__bf16*)(smem + 98304);               // 32KB
  __bf16* act2 = (__bf16*)(smem + 98304 + 32768);       // 32KB
  float* stageF = (float*)(smem + 98304);               // 64KB TDM staging (pre-MLP only)
  const int tid = threadIdx.x, lane = tid & 31, wave = tid >> 5;
  const long r0 = (long)blockIdx.x * ROWS;

  // async DMA of the contiguous h_E tile into LDS (one issue, wave 0)
  if (wave == 0) tdm_load_f32(98304u, hE + r0 * Hdim, ROWS * Hdim);

  // meanwhile stage hV / neighbor columns as bf16
  for (int t = tid; t < ROWS * 256; t += 256) {
    int row = t >> 8, cc = t & 255;
    long r = r0 + row;
    long bl = r / Kdim;                                  // b*L + l
    float x;
    int c;
    if (cc < Hdim) {
      c = cc;
      x = ldgf(hV + bl * Hdim + cc);
    } else {
      c = cc + Hdim;
      int nb = ldgi(Eidx + r);
      long b = bl / Ldim;
      x = ldgf(hV + (b * Ldim + nb) * Hdim + (cc - Hdim));
    }
    Abuf[row * KIN + c] = (__bf16)x;
  }
  if (wave == 0) __builtin_amdgcn_s_wait_tensorcnt(0);
  __syncthreads();
  for (int t = tid; t < ROWS * Hdim; t += 256) {
    int row = t >> 7, c = t & 127;
    Abuf[row * KIN + Hdim + c] = (__bf16)stageF[t];
  }
  __syncthreads();

  v8f out[8];
  mlp3_core<KIN>(Abuf, w1t, b1, w2t, b2, w3t, act1, act2, out, lane, wave, false);

  // epilogue: +b3, * mask_attend, atomic scatter-sum over k into dh[b,l,:]
  const int nidx = lane & 15, g = lane >> 4, m0 = wave * 16;
  float mk[8];
  int bl8[8];
#pragma unroll
  for (int j = 0; j < 8; ++j) {
    long r = r0 + m0 + j + g * 8;
    mk[j] = ldgf(mask_attend + r);
    bl8[j] = (int)(r / Kdim);
  }
#pragma unroll
  for (int nt = 0; nt < 8; ++nt) {
    int col = nt * 16 + nidx;
    float bcv = ldgf(b3 + col);
#pragma unroll
    for (int j = 0; j < 8; ++j)
      atomicAdd(dh + (size_t)bl8[j] * Hdim + col, (out[nt][j] + bcv) * mk[j]);
  }
}

// ---------------- kernel 2: residual+LN1, FFN (WMMA), residual+LN2, mask ----------------
__global__ __launch_bounds__(256) void k_node_ffn(
    const float* __restrict__ hV, const float* __restrict__ dh,
    const float* __restrict__ g1, const float* __restrict__ be1,
    const __bf16* __restrict__ wit, const float* __restrict__ bi,
    const __bf16* __restrict__ wot, const float* __restrict__ bo,
    const float* __restrict__ g2, const float* __restrict__ be2,
    const float* __restrict__ mask_V, float* __restrict__ outV) {
  extern __shared__ char smem[];
  float*  xbuf = (float*)smem;                          // 128*128*4 = 64KB
  __bf16* Abuf = (__bf16*)(smem + 65536);               // 32KB
  __bf16* act  = (__bf16*)(smem + 98304);               // 128*512*2 = 128KB
  const int tid = threadIdx.x, lane = tid & 31, wave = tid >> 5;
  const long p0 = (long)blockIdx.x * 128;

  // DMA h_V tile straight into xbuf, then add dh/SCALE
  if (wave == 0) {
    tdm_load_f32(0u, hV + p0 * Hdim, 128 * Hdim);
    __builtin_amdgcn_s_wait_tensorcnt(0);
  }
  __syncthreads();
  for (int t = tid; t < 128 * Hdim; t += 256) {
    int row = t >> 7, c = t & 127;
    long p = p0 + row;
    xbuf[t] += ldgf(dh + p * Hdim + c) * SCALE_INV;
  }
  __syncthreads();

  // LN1 (row per thread); normalized f32 stays in xbuf, bf16 copy in Abuf
  if (tid < 128) {
    int row = tid;
    float mean = 0.f;
    for (int c = 0; c < Hdim; ++c) mean += xbuf[row * Hdim + c];
    mean *= (1.0f / Hdim);
    float var = 0.f;
    for (int c = 0; c < Hdim; ++c) {
      float d = xbuf[row * Hdim + c] - mean;
      var += d * d;
    }
    var *= (1.0f / Hdim);
    float rs = rsqrtf(var + LN_EPS);
    for (int c = 0; c < Hdim; ++c) {
      float nrm = (xbuf[row * Hdim + c] - mean) * rs * ldgf(g1 + c) + ldgf(be1 + c);
      xbuf[row * Hdim + c] = nrm;
      Abuf[row * Hdim + c] = (__bf16)nrm;
    }
  }
  __syncthreads();

  const int m0 = wave * 16, nidx = lane & 15, g = lane >> 4;

  // FFN layer 1: [128x128] @ wi -> gelu -> act [128x512]
  v16bf af[4];
#pragma unroll
  for (int kt = 0; kt < 4; ++kt) af[kt] = load_a_frag(Abuf, m0, kt * 32, Hdim, lane);
  for (int nt = 0; nt < 32; ++nt) {
    v8f acc = {};
    v16bf bc_ = load_b_frag(wit, nt * 16, 0, Hdim, lane);
#pragma unroll
    for (int kt = 0; kt < 4; ++kt) {
      v16bf bn = bc_;
      if (kt + 1 < 4) bn = load_b_frag(wit, nt * 16, (kt + 1) * 32, Hdim, lane);
      acc = wmma_bf16(af[kt], bc_, acc);
      bc_ = bn;
    }
    int col = nt * 16 + nidx;
    float bcv = ldgf(bi + col);
#pragma unroll
    for (int j = 0; j < 8; ++j) {
      int row = m0 + j + g * 8;
      act[row * FFdim + col] = (__bf16)gelu_f(acc[j] + bcv);
    }
  }

  // FFN layer 2: act @ wo + residual into xbuf
#pragma unroll
  for (int nt = 0; nt < 8; ++nt) {
    v8f acc = {};
    v16bf bc_ = load_b_frag(wot, nt * 16, 0, FFdim, lane);
    for (int kt = 0; kt < 16; ++kt) {
      v16bf bn = bc_;
      if (kt + 1 < 16) bn = load_b_frag(wot, nt * 16, (kt + 1) * 32, FFdim, lane);
      acc = wmma_bf16(load_a_frag(act, m0, kt * 32, FFdim, lane), bc_, acc);
      bc_ = bn;
    }
    int col = nt * 16 + nidx;
    float bcv = ldgf(bo + col);
#pragma unroll
    for (int j = 0; j < 8; ++j) {
      int row = m0 + j + g * 8;
      xbuf[row * Hdim + col] += acc[j] + bcv;
    }
  }
  __syncthreads();

  // LN2 + mask_V -> d_out (node part)
  if (tid < 128) {
    int row = tid;
    long p = p0 + row;
    float mean = 0.f;
    for (int c = 0; c < Hdim; ++c) mean += xbuf[row * Hdim + c];
    mean *= (1.0f / Hdim);
    float var = 0.f;
    for (int c = 0; c < Hdim; ++c) {
      float d = xbuf[row * Hdim + c] - mean;
      var += d * d;
    }
    var *= (1.0f / Hdim);
    float rs = rsqrtf(var + LN_EPS);
    float mv = ldgf(mask_V + p);
    for (int c = 0; c < Hdim; ++c)
      outV[p * Hdim + c] = mv * ((xbuf[row * Hdim + c] - mean) * rs * ldgf(g2 + c) + ldgf(be2 + c));
  }
}

// ---------------- kernel 3: edge-update MLP + residual + LN3 ----------------
__global__ __launch_bounds__(256) void k_edge_edge(
    const float* __restrict__ hVnew, const float* __restrict__ hE,
    const int* __restrict__ Eidx,
    const __bf16* __restrict__ w11t, const float* __restrict__ b11,
    const __bf16* __restrict__ w12t, const float* __restrict__ b12,
    const __bf16* __restrict__ w13t, const float* __restrict__ b13,
    const float* __restrict__ g3, const float* __restrict__ be3,
    float* __restrict__ outE) {
  extern __shared__ char smem[];
  __bf16* Abuf = (__bf16*)smem;                         // 96KB (reused as outf)
  __bf16* act1 = (__bf16*)(smem + 98304);
  __bf16* act2 = (__bf16*)(smem + 98304 + 32768);
  float* stageF = (float*)(smem + 98304);               // 64KB TDM staging (pre-MLP only)
  float*  outf = (float*)smem;                          // 64KB, valid after mlp3 syncA
  const int tid = threadIdx.x, lane = tid & 31, wave = tid >> 5;
  const long r0 = (long)blockIdx.x * ROWS;

  if (wave == 0) tdm_load_f32(98304u, hE + r0 * Hdim, ROWS * Hdim);

  for (int t = tid; t < ROWS * 256; t += 256) {
    int row = t >> 8, cc = t & 255;
    long r = r0 + row;
    long bl = r / Kdim;
    float x;
    int c;
    if (cc < Hdim) {
      c = cc;
      x = ldgf(hVnew + bl * Hdim + cc);
    } else {
      c = cc + Hdim;
      int nb = ldgi(Eidx + r);
      long b = bl / Ldim;
      x = ldgf(hVnew + (b * Ldim + nb) * Hdim + (cc - Hdim));
    }
    Abuf[row * KIN + c] = (__bf16)x;
  }
  if (wave == 0) __builtin_amdgcn_s_wait_tensorcnt(0);
  __syncthreads();
  for (int t = tid; t < ROWS * Hdim; t += 256) {
    int row = t >> 7, c = t & 127;
    Abuf[row * KIN + Hdim + c] = (__bf16)stageF[t];
  }
  __syncthreads();

  v8f out[8];
  // syncA=true: barrier after A fragments consumed so outf may reuse Abuf region
  mlp3_core<KIN>(Abuf, w11t, b11, w12t, b12, w13t, act1, act2, out, lane, wave, true);

  // epilogue: msg + b13 + h_E residual -> LDS f32 tile
  const int nidx = lane & 15, g = lane >> 4, m0 = wave * 16;
#pragma unroll
  for (int nt = 0; nt < 8; ++nt) {
    int col = nt * 16 + nidx;
    float bcv = ldgf(b13 + col);
#pragma unroll
    for (int j = 0; j < 8; ++j) {
      int row = m0 + j + g * 8;
      long r = r0 + row;
      outf[row * Hdim + col] = ldgf(hE + r * Hdim + col) + out[nt][j] + bcv;
    }
  }
  __syncthreads();

  // LN3 per row -> d_out (edge part)
  if (tid < 128) {
    int row = tid;
    long r = r0 + row;
    float mean = 0.f;
    for (int c = 0; c < Hdim; ++c) mean += outf[row * Hdim + c];
    mean *= (1.0f / Hdim);
    float var = 0.f;
    for (int c = 0; c < Hdim; ++c) {
      float d = outf[row * Hdim + c] - mean;
      var += d * d;
    }
    var *= (1.0f / Hdim);
    float rs = rsqrtf(var + LN_EPS);
    for (int c = 0; c < Hdim; ++c)
      outE[r * Hdim + c] = (outf[row * Hdim + c] - mean) * rs * ldgf(g3 + c) + ldgf(be3 + c);
  }
}

// ---------------- launch ----------------
extern "C" void kernel_launch(void* const* d_in, const int* in_sizes, int n_in,
                              void* d_out, int out_size, void* d_ws, size_t ws_size,
                              hipStream_t stream) {
  const float* hV   = (const float*)d_in[0];
  const float* hE   = (const float*)d_in[1];
  const float* mV   = (const float*)d_in[2];
  const float* mAtt = (const float*)d_in[3];
  const int*   Eidx = (const int*)d_in[4];
  const float* w1  = (const float*)d_in[5];  const float* b1  = (const float*)d_in[6];
  const float* w2  = (const float*)d_in[7];  const float* b2  = (const float*)d_in[8];
  const float* w3  = (const float*)d_in[9];  const float* b3  = (const float*)d_in[10];
  const float* w11 = (const float*)d_in[11]; const float* b11 = (const float*)d_in[12];
  const float* w12 = (const float*)d_in[13]; const float* b12 = (const float*)d_in[14];
  const float* w13 = (const float*)d_in[15]; const float* b13 = (const float*)d_in[16];
  const float* g1  = (const float*)d_in[17]; const float* be1 = (const float*)d_in[18];
  const float* g2  = (const float*)d_in[19]; const float* be2 = (const float*)d_in[20];
  const float* g3  = (const float*)d_in[21]; const float* be3 = (const float*)d_in[22];
  const float* wi  = (const float*)d_in[23]; const float* bi  = (const float*)d_in[24];
  const float* wo  = (const float*)d_in[25]; const float* bo  = (const float*)d_in[26];

  const int BLH  = Bdim * Ldim * Hdim;            // 524288
  const long BLKH = (long)Bdim * Ldim * Kdim * Hdim;

  // workspace layout: [dh f32 BLH][bf16 transposed weights]
  float*  dh = (float*)d_ws;
  __bf16* wsp = (__bf16*)((char*)d_ws + (size_t)BLH * 4);
  __bf16* w1t  = wsp;                 // 128*384
  __bf16* w2t  = w1t  + 49152;        // 128*128
  __bf16* w3t  = w2t  + 16384;
  __bf16* w11t = w3t  + 16384;        // 128*384
  __bf16* w12t = w11t + 49152;
  __bf16* w13t = w12t + 16384;
  __bf16* wit  = w13t + 16384;        // 512*128
  __bf16* wot  = wit  + 65536;        // 128*512

  float* outV = (float*)d_out;                    // [B,L,H]
  float* outE = (float*)d_out + BLH;              // [B,L,K,H]

  // prep: zero dh + convert/transpose weights to bf16 [N][K]
  k_zero<<<(BLH + 255) / 256, 256, 0, stream>>>(dh, BLH);
  k_wt<<<(KIN * Hdim + 255) / 256, 256, 0, stream>>>(w1,  w1t,  KIN,  Hdim);
  k_wt<<<(Hdim * Hdim + 255) / 256, 256, 0, stream>>>(w2,  w2t,  Hdim, Hdim);
  k_wt<<<(Hdim * Hdim + 255) / 256, 256, 0, stream>>>(w3,  w3t,  Hdim, Hdim);
  k_wt<<<(KIN * Hdim + 255) / 256, 256, 0, stream>>>(w11, w11t, KIN,  Hdim);
  k_wt<<<(Hdim * Hdim + 255) / 256, 256, 0, stream>>>(w12, w12t, Hdim, Hdim);
  k_wt<<<(Hdim * Hdim + 255) / 256, 256, 0, stream>>>(w13, w13t, Hdim, Hdim);
  k_wt<<<(Hdim * FFdim + 255) / 256, 256, 0, stream>>>(wi, wit, Hdim, FFdim);
  k_wt<<<(FFdim * Hdim + 255) / 256, 256, 0, stream>>>(wo, wot, FFdim, Hdim);

  const int edgeBlocks = (int)(BLKH / Hdim / ROWS);   // 1536
  const size_t edgeLds = 98304 + 32768 + 32768;       // 160KB
  k_edge_node<<<edgeBlocks, 256, edgeLds, stream>>>(hV, hE, mAtt, Eidx,
                                                    w1t, b1, w2t, b2, w3t, b3, dh);

  const int nodeBlocks = (Bdim * Ldim) / 128;          // 32
  const size_t nodeLds = 65536 + 32768 + 131072;       // 224KB
  k_node_ffn<<<nodeBlocks, 256, nodeLds, stream>>>(hV, dh, g1, be1, wit, bi, wot, bo,
                                                   g2, be2, mV, outV);

  k_edge_edge<<<edgeBlocks, 256, edgeLds, stream>>>(outV, hE, Eidx,
                                                    w11t, b11, w12t, b12, w13t, b13,
                                                    g3, be3, outE);
}
